// CumulativeFlattenedLinear_19490561589875
// MI455X (gfx1250) — matrix operand order repaired and checked by reading, not compiled, not run on hardware
//
#include <hip/hip_runtime.h>
#include <hip/hip_bf16.h>

typedef _Float16 v16h __attribute__((ext_vector_type(16)));
typedef _Float16 v8h  __attribute__((ext_vector_type(8)));
typedef _Float16 v4h  __attribute__((ext_vector_type(4)));
typedef float    v8f  __attribute__((ext_vector_type(8)));

#define B_    16
#define C_    64
#define T_    8192
#define O_    256
#define KK    32
#define CK    (C_*KK)        // 2048
#define PADL  39             // K-1 + n_discard
#define SLICE 96             // staged samples per channel (95 used, padded)

// workspace layout (bytes):
//   0:  u32 wmax_bits, 4: u32 bmax_bits, 8: u32 ymax_bits
//   1024: float qb[O_]
//   4096: _Float16 qw[O_*CK]   (1 MB)
#define WS_QB_OFF 1024
#define WS_QW_OFF 4096

__global__ void k_init(unsigned* u) {
    if (threadIdx.x < 3) u[threadIdx.x] = 0u;
}

__global__ void k_absmax(const float* __restrict__ w, int n, unsigned* __restrict__ target) {
    __shared__ float sm[256];
    float m = 0.f;
    for (int i = blockIdx.x * blockDim.x + threadIdx.x; i < n; i += gridDim.x * blockDim.x)
        m = fmaxf(m, fabsf(w[i]));
    sm[threadIdx.x] = m;
    __syncthreads();
    for (int s = 128; s > 0; s >>= 1) {
        if ((int)threadIdx.x < s) sm[threadIdx.x] = fmaxf(sm[threadIdx.x], sm[threadIdx.x + s]);
        __syncthreads();
    }
    if (threadIdx.x == 0) atomicMax(target, __float_as_uint(sm[0]));
}

__global__ void k_quant(const float* __restrict__ w, const float* __restrict__ bias,
                        const unsigned* __restrict__ u, float* __restrict__ qb,
                        _Float16* __restrict__ qw) {
    float wmax = __uint_as_float(u[0]);
    float wq   = ceilf(log2f(wmax + 1e-12f)) - 7.f;   // BITS_W-1 = 7
    float wsc  = exp2f(wq), wsi = exp2f(-wq);
    int idx    = blockIdx.x * blockDim.x + threadIdx.x;
    int stride = gridDim.x * blockDim.x;
    for (int i = idx; i < O_ * CK; i += stride) {
        float q = rintf(w[i] * wsi);                  // round-half-even like jnp.round
        q = fminf(fmaxf(q, -128.f), 127.f);
        qw[i] = (_Float16)(q * wsc);                  // int8 * 2^q : exact in f16
    }
    if (idx < O_) {
        float bmax = __uint_as_float(u[1]);
        float bq   = ceilf(log2f(bmax + 1e-12f)) - 15.f;  // BITS_A-1 = 15
        float bs   = exp2f(bq);
        float q    = rintf(bias[idx] / bs);
        q = fminf(fmaxf(q, -32768.f), 32767.f);
        qb[idx] = q * bs;
    }
}

// Main GEMM: D[O=256, N=B*T] = QW[256,2048] x Window(x)[2048, B*T]
// workgroup: 256 threads = 8 waves; tile M=256 x N=64 (one b, 64 consecutive t)
// wave tile: M=64 x N=32 -> 4x2 accumulator tiles of 16x16 f32
__global__ void __launch_bounds__(256)
k_gemm(const float* __restrict__ x, const _Float16* __restrict__ qw,
       const float* __restrict__ qb, float* __restrict__ y,
       unsigned* __restrict__ ymax_bits) {
    // x slice staged as 4-wide redundant half slots: slot[c][i] = {x[i],x[i+1],x[i+2],x[i+3]}
    __shared__ __align__(16) _Float16 sh_hi[32 * SLICE * 4];  // 24 KB
    __shared__ __align__(16) _Float16 sh_lo[32 * SLICE * 4];  // 24 KB
    __shared__ float smax[256];

    const int tid    = threadIdx.x;
    const int lane   = tid & 31;
    const int wv     = tid >> 5;      // wave id 0..7
    const int m_base = (wv & 3) * 64; // 4 waves stacked in M
    const int n_base = (wv >> 2) * 32;// 2 waves stacked in N
    const int b      = blockIdx.x >> 7;
    const int t0     = (blockIdx.x & 127) << 6;

    const int l15    = lane & 15;
    const int hiHalf = lane >> 4;     // 0 for lanes 0-15, 1 for 16-31
    const int aksel  = hiHalf * 16;   // A frag: K 0..15 vs 16..31
    const int bk0    = hiHalf * 16;   // B frag: K 0..15 vs 16..31

    // per-lane weight row base pointers (4 M-subtiles)
    const _Float16* wbase[4];
    #pragma unroll
    for (int mi = 0; mi < 4; ++mi)
        wbase[mi] = qw + (size_t)(m_base + mi * 16 + l15) * CK + aksel;

    // per-lane B-fragment LDS half-offset base (2 N-subtiles)
    int bso[2];
    #pragma unroll
    for (int ni = 0; ni < 2; ++ni)
        bso[ni] = (n_base + ni * 16 + l15 + bk0) * 4;

    // A-fragment loader (global, L2-resident f16 weights)
    auto loadA = [&](int c, v16h* dst) {
        const int kb = c * KK;
        #pragma unroll
        for (int mi = 0; mi < 4; ++mi) {
            const _Float16* wp = wbase[mi] + kb;
            v8h a0 = *(const v8h*)(wp);        // K kb+sel+0..7
            v8h a1 = *(const v8h*)(wp + 8);    // K kb+sel+8..15
            dst[mi] = __builtin_shufflevector(a0, a1,
                0,1,2,3,4,5,6,7,8,9,10,11,12,13,14,15);
        }
    };

    // B-fragment loader (LDS hi/lo slot arrays)
    auto loadB = [&](int cc, v16h* bhi, v16h* blo) {
        const int sbase = cc * SLICE * 4;
        #pragma unroll
        for (int ni = 0; ni < 2; ++ni) {
            int so = sbase + bso[ni];
            v4h h0 = *(const v4h*)&sh_hi[so];
            v4h h1 = *(const v4h*)&sh_hi[so + 16];
            v4h h2 = *(const v4h*)&sh_hi[so + 32];
            v4h h3 = *(const v4h*)&sh_hi[so + 48];
            v8h ph0 = __builtin_shufflevector(h0, h1, 0,1,2,3,4,5,6,7);
            v8h ph1 = __builtin_shufflevector(h2, h3, 0,1,2,3,4,5,6,7);
            bhi[ni] = __builtin_shufflevector(ph0, ph1,
                0,1,2,3,4,5,6,7,8,9,10,11,12,13,14,15);
            v4h l0 = *(const v4h*)&sh_lo[so];
            v4h l1 = *(const v4h*)&sh_lo[so + 16];
            v4h l2 = *(const v4h*)&sh_lo[so + 32];
            v4h l3 = *(const v4h*)&sh_lo[so + 48];
            v8h pl0 = __builtin_shufflevector(l0, l1, 0,1,2,3,4,5,6,7);
            v8h pl1 = __builtin_shufflevector(l2, l3, 0,1,2,3,4,5,6,7);
            blo[ni] = __builtin_shufflevector(pl0, pl1,
                0,1,2,3,4,5,6,7,8,9,10,11,12,13,14,15);
        }
    };

    // 16 WMMAs for one K-step (4M x 2N x {hi,lo})
    auto mma = [&](v8f (&ac)[4][2], const v16h* a, const v16h* bhi, const v16h* blo) {
        #pragma unroll
        for (int mi = 0; mi < 4; ++mi) {
            #pragma unroll
            for (int ni = 0; ni < 2; ++ni) {
                ac[mi][ni] = __builtin_amdgcn_wmma_f32_16x16x32_f16(
                    false, a[mi], false, bhi[ni], (short)0, ac[mi][ni], false, false);
                ac[mi][ni] = __builtin_amdgcn_wmma_f32_16x16x32_f16(
                    false, a[mi], false, blo[ni], (short)0, ac[mi][ni], false, false);
            }
        }
    };

    v8f acc[4][2] = {};
    v16h aA[4], aB[4];          // named ping-pong buffers: no runtime indexing
    v16h bhi[2], blo[2];

    for (int chunk = 0; chunk < 2; ++chunk) {
        __syncthreads();
        // --- stage 32 channels of x, hi/lo f16 split, 4-wide redundant slots ---
        for (int idx = tid; idx < 32 * SLICE; idx += 256) {
            int cc  = idx / SLICE, i = idx - cc * SLICE;
            int c   = chunk * 32 + cc;
            const float* xc = x + ((size_t)(b * C_ + c) * T_);
            int tau = t0 - PADL + i;
            _Float16 h[4], l[4];
            #pragma unroll
            for (int j = 0; j < 4; ++j) {
                int tj  = tau + j;
                float v = (tj >= 0 && tj < T_) ? xc[tj] : 0.f;
                _Float16 hv = (_Float16)v;
                h[j] = hv;
                l[j] = (_Float16)(v - (float)hv);
            }
            int so = (cc * SLICE + i) * 4;
            v4h vh = {h[0], h[1], h[2], h[3]};
            v4h vl = {l[0], l[1], l[2], l[3]};
            *(v4h*)&sh_hi[so] = vh;
            *(v4h*)&sh_lo[so] = vl;
        }
        __syncthreads();

        // prefetch next chunk's x slice into cache while this chunk computes
        if (chunk == 0) {
            for (int idx = tid; idx < 32 * SLICE; idx += 256) {
                int cc = idx / SLICE, i = idx - cc * SLICE;
                if ((i & 31) == 0) {                      // one per 128B line
                    int c   = 32 + cc;
                    int tau = t0 - PADL + i;
                    int tc  = tau < 0 ? 0 : (tau >= T_ ? T_ - 1 : tau);
                    __builtin_prefetch(x + ((size_t)(b * C_ + c) * T_) + tc, 0, 3);
                }
            }
        }

        // --- K loop: one channel per 32-wide K step, 2 steps/iter, ping-pong A ---
        const int c0 = chunk * 32;
        loadA(c0, aA);
        #pragma unroll 1
        for (int cc = 0; cc < 32; cc += 2) {
            // stream-prefetch weights ~8 K-steps (512B/row) ahead
            #pragma unroll
            for (int mi = 0; mi < 4; ++mi)
                __builtin_prefetch(wbase[mi] + (c0 + cc) * KK + 8 * KK, 0, 3);

            // phase 0: prefetch A for cc+1, compute with aA
            loadA(c0 + cc + 1, aB);
            loadB(cc, bhi, blo);
            mma(acc, aA, bhi, blo);

            // phase 1: prefetch A for cc+2, compute with aB
            if (cc + 2 < 32) loadA(c0 + cc + 2, aA);
            loadB(cc + 1, bhi, blo);
            mma(acc, aB, bhi, blo);
        }
    }

    // --- epilogue: bias add, store, block-level |y| max ---
    float lmax = 0.f;
    #pragma unroll
    for (int mi = 0; mi < 4; ++mi) {
        #pragma unroll
        for (int ni = 0; ni < 2; ++ni) {
            int o0 = m_base + mi * 16 + hiHalf * 8;      // C/D layout: lanes>=16 hold M=8..15
            int t  = t0 + n_base + ni * 16 + l15;
            float* yp = y + ((size_t)(b * O_ + o0) * T_) + t;
            #pragma unroll
            for (int j = 0; j < 8; ++j) {
                float v = acc[mi][ni][j] + qb[o0 + j];
                yp[(size_t)j * T_] = v;
                lmax = fmaxf(lmax, fabsf(v));
            }
        }
    }
    smax[tid] = lmax;
    __syncthreads();
    for (int s = 128; s > 0; s >>= 1) {
        if (tid < s) smax[tid] = fmaxf(smax[tid], smax[tid + s]);
        __syncthreads();
    }
    if (tid == 0) atomicMax(ymax_bits, __float_as_uint(smax[0]));
}

// Output requant with rounded=False and pow2 scale == clamp to [-32768*s, 32767*s]
__global__ void k_requant(float* __restrict__ y, const unsigned* __restrict__ u, size_t n) {
    float ym = __uint_as_float(u[2]);
    float q  = ceilf(log2f(ym + 1e-12f)) - 15.f;
    float s  = exp2f(q);
    float hi = 32767.f * s, lo = -32768.f * s;
    for (size_t i = blockIdx.x * (size_t)blockDim.x + threadIdx.x; i < n;
         i += (size_t)gridDim.x * blockDim.x) {
        float v = y[i];
        y[i] = fminf(fmaxf(v, lo), hi);
    }
}

extern "C" void kernel_launch(void* const* d_in, const int* in_sizes, int n_in,
                              void* d_out, int out_size, void* d_ws, size_t ws_size,
                              hipStream_t stream) {
    const float* x    = (const float*)d_in[0];   // [16, 64, 8192]
    const float* w    = (const float*)d_in[1];   // [256, 2048]
    const float* bias = (const float*)d_in[2];   // [256]
    // d_in[3] = n_discard (== 8, baked into PADL)

    unsigned*  u  = (unsigned*)d_ws;
    float*     qb = (float*)((char*)d_ws + WS_QB_OFF);
    _Float16*  qw = (_Float16*)((char*)d_ws + WS_QW_OFF);
    float*     y  = (float*)d_out;

    k_init<<<1, 32, 0, stream>>>(u);
    k_absmax<<<256, 256, 0, stream>>>(w, O_ * CK, u + 0);
    k_absmax<<<1, 256, 0, stream>>>(bias, O_, u + 1);
    k_quant<<<512, 256, 0, stream>>>(w, bias, u, qb, qw);
    k_gemm<<<(B_ * T_) / 64, 256, 0, stream>>>(x, qw, qb, y, u + 2);
    size_t n = (size_t)B_ * O_ * T_;
    k_requant<<<4096, 256, 0, stream>>>(y, u, n);
}
